// MultiHeadPAATLayer_10093173145793
// MI455X (gfx1250) — compile-verified, matrix operand order checked by reading, not run on hardware
//
#include <hip/hip_runtime.h>
#include <hip/hip_bf16.h>

// ---------------------------------------------------------------------------
// Multi-head GAT layer for MI455X (gfx1250, wave32, WMMA)
//   N=50000 nodes, E=800000 edges, IN=256, OUT=64, H=4  (C = H*OUT = 256)
// Pipeline:
//   1) h (f32) -> hh (f16)                              [elementwise]
//   2) W (H,IN,OUT) -> Wt (f16, [c][k] col-major)       [tiny]
//   3) z = hh @ Wcat  via v_wmma_f32_16x16x32_f16       [compute kernel]
//   4) s1[h][n] = z[n,h*64:]·a[h,:64], s2 likewise      [per-node scalars]
//   5) row_ptr via binary search (dst is sorted)        [CSR, atomic-free]
//   6) per-(node,head) wave: softmax-max, then fused
//      exp-weighted gather-accumulate of z[src], /denom [bandwidth limiter]
// ---------------------------------------------------------------------------

typedef __attribute__((ext_vector_type(16))) _Float16 v16h;
typedef __attribute__((ext_vector_type(8)))  _Float16 v8h;
typedef __attribute__((ext_vector_type(8)))  float    v8f;

#define IN_DIM  256
#define OUT_DIM 64
#define N_HEADS 4
#define C_DIM   256   // N_HEADS * OUT_DIM

// ---- 1) f32 -> f16 convert ------------------------------------------------
__global__ void cvt_h_kernel(const float* __restrict__ h, _Float16* __restrict__ hh, int n) {
    int i = blockIdx.x * blockDim.x + threadIdx.x;
    if (i < n) hh[i] = (_Float16)h[i];
}

// ---- 2) build Wt[c][k] = W[h][k][o]  (c = h*64+o), f16 --------------------
__global__ void build_wt_kernel(const float* __restrict__ W, _Float16* __restrict__ Wt) {
    int c = blockIdx.x;        // 0..255
    int k = threadIdx.x;       // 0..255
    int hd = c >> 6, o = c & 63;
    Wt[c * IN_DIM + k] = (_Float16)W[(size_t)hd * IN_DIM * OUT_DIM + (size_t)k * OUT_DIM + o];
}

// ---- 3) GEMM: Z[M x 256] = hh[M x 256] * Wcat[256 x 256] ------------------
// Block: 256 threads = 8 waves arranged 2(row) x 4(col): 32x64 block tile.
// Per wave: one 16x16 f32 accumulator, K looped in steps of 32.
// Fragment layout per CDNA5 ISA (16-bit A 16x32): lane p=lane>>4, m=lane&15,
// halves 0..7 = K kk+p*8.., halves 8..15 = K kk+16+p*8..  (two 16B loads).
// B uses same per-lane packing indexed by column (Wt stored [c][k]).
union V16 { v16h v; v8h h[2]; };

__global__ void __launch_bounds__(256)
gemm_wmma_kernel(const _Float16* __restrict__ A, const _Float16* __restrict__ Bt,
                 float* __restrict__ Z, int M) {
    const int K = IN_DIM;
    int tid  = threadIdx.x;
    int wave = tid >> 5;
    int lane = tid & 31;
    int wm = wave & 1;          // 0..1
    int wn = wave >> 1;         // 0..3
    int row0 = blockIdx.x * 32 + wm * 16;
    int col0 = blockIdx.y * 64 + wn * 16;
    int p  = lane >> 4;         // half-wave select
    int mr = lane & 15;

    int arow = row0 + mr; if (arow >= M) arow = M - 1;   // clamp (store is guarded)
    int bcol = col0 + mr;
    const _Float16* Ap = A  + (size_t)arow * K;
    const _Float16* Bp = Bt + (size_t)bcol * K;

    v8f acc = {};
    #pragma unroll
    for (int kk = 0; kk < K; kk += 32) {
        V16 a, b;
        a.h[0] = *(const v8h*)(Ap + kk + p * 8);
        a.h[1] = *(const v8h*)(Ap + kk + 16 + p * 8);
        b.h[0] = *(const v8h*)(Bp + kk + p * 8);
        b.h[1] = *(const v8h*)(Bp + kk + 16 + p * 8);
        acc = __builtin_amdgcn_wmma_f32_16x16x32_f16(
                  /*neg_a=*/false, a.v, /*neg_b=*/false, b.v,
                  /*c_mod=*/(short)0, acc, /*reuse_a=*/false, /*reuse_b=*/false);
    }
    // D layout: VGPR r -> M = p*8 + r, N = mr
    #pragma unroll
    for (int r = 0; r < 8; ++r) {
        int orow = row0 + p * 8 + r;
        if (orow < M) Z[(size_t)orow * C_DIM + col0 + mr] = acc[r];
    }
}

// ---- 4) per-node attention scalars ----------------------------------------
// s1[h*N+n] = sum_o z[n][h*64+o]*a[h][o] ; s2 uses a[h][64+o]
__global__ void attn_scal_kernel(const float* __restrict__ z, const float* __restrict__ a,
                                 float* __restrict__ s1, float* __restrict__ s2, int N) {
    int t = blockIdx.x * blockDim.x + threadIdx.x;
    if (t >= N * N_HEADS) return;
    int n = t >> 2, hd = t & 3;
    const float* zp = z + (size_t)n * C_DIM + hd * OUT_DIM;
    const float* a1 = a + hd * (2 * OUT_DIM);
    const float* a2 = a1 + OUT_DIM;
    float acc1 = 0.f, acc2 = 0.f;
    #pragma unroll 8
    for (int o = 0; o < OUT_DIM; ++o) {
        float zv = zp[o];
        acc1 = fmaf(zv, a1[o], acc1);
        acc2 = fmaf(zv, a2[o], acc2);
    }
    s1[(size_t)hd * N + n] = acc1;
    s2[(size_t)hd * N + n] = acc2;
}

// ---- 5) CSR row pointers from sorted dst ----------------------------------
__global__ void rowptr_kernel(const int* __restrict__ dst, int E, int N, int* __restrict__ row_ptr) {
    int n = blockIdx.x * blockDim.x + threadIdx.x;
    if (n > N) return;
    int lo = 0, hi = E;                       // lower_bound(dst, n)
    while (lo < hi) { int mid = (lo + hi) >> 1; if (dst[mid] < n) lo = mid + 1; else hi = mid; }
    row_ptr[n] = lo;
}

// ---- 6) per-(node,head) softmax aggregation (one wave32 per head) ---------
__global__ void __launch_bounds__(128)
gat_aggregate_kernel(const float* __restrict__ z, const float* __restrict__ s1,
                     const float* __restrict__ s2, const int* __restrict__ src,
                     const int* __restrict__ row_ptr, float* __restrict__ out, int N) {
    int n    = blockIdx.x;
    int hd   = threadIdx.x >> 5;   // wave index == head
    int lane = threadIdx.x & 31;
    int e0 = row_ptr[n], e1 = row_ptr[n + 1];

    float acc0 = 0.f, acc1 = 0.f;
    if (e0 < e1) {
        const float* s1h = s1 + (size_t)hd * N;
        float s2n = s2[(size_t)hd * N + n];

        // pass 1: wave-parallel max of leaky_relu scores
        float m = -3.402823466e38f;
        for (int e = e0 + lane; e < e1; e += 32) {
            float v = s1h[src[e]] + s2n;
            v = v > 0.f ? v : 0.01f * v;
            m = fmaxf(m, v);
        }
        #pragma unroll
        for (int off = 16; off > 0; off >>= 1)
            m = fmaxf(m, __shfl_xor(m, off, 32));

        // pass 2: fused exp-weight + gather-accumulate (2 coalesced 128B loads/edge)
        float denom = 0.f;
        for (int e = e0; e < e1; ++e) {
            int s = src[e];                        // uniform across wave -> broadcast
            float v = s1h[s] + s2n;
            v = v > 0.f ? v : 0.01f * v;
            float w = __expf(v - m);
            denom += w;
            const float* zp = z + (size_t)s * C_DIM + hd * OUT_DIM;
            acc0 = fmaf(w, zp[lane],      acc0);
            acc1 = fmaf(w, zp[lane + 32], acc1);
        }
        float inv = 1.0f / denom;                  // denom >= 1 (argmax edge has w==1)
        acc0 *= inv; acc1 *= inv;
    }
    out[(size_t)n * C_DIM + hd * OUT_DIM + lane]      = acc0;
    out[(size_t)n * C_DIM + hd * OUT_DIM + lane + 32] = acc1;
}

// ---------------------------------------------------------------------------
extern "C" void kernel_launch(void* const* d_in, const int* in_sizes, int n_in,
                              void* d_out, int out_size, void* d_ws, size_t ws_size,
                              hipStream_t stream) {
    const float* h   = (const float*)d_in[0];   // (N, 256)
    const float* W   = (const float*)d_in[1];   // (4, 256, 64)
    const float* a   = (const float*)d_in[2];   // (4, 128)
    const int*   src = (const int*)  d_in[3];   // (E,)
    const int*   dst = (const int*)  d_in[4];   // (E,) sorted
    float*       out = (float*)d_out;           // (N, 256)

    const int N = in_sizes[0] / IN_DIM;
    const int E = in_sizes[3];

    // workspace carve-up (256B aligned)
    char* ws = (char*)d_ws;
    size_t off = 0;
    auto carve = [&](size_t bytes) -> char* {
        char* p = ws + off;
        off += (bytes + 255) & ~(size_t)255;
        return p;
    };
    _Float16* hh      = (_Float16*)carve((size_t)N * IN_DIM * sizeof(_Float16)); // 25.6 MB
    _Float16* Wt      = (_Float16*)carve((size_t)C_DIM * IN_DIM * sizeof(_Float16));
    float*    z       = (float*)   carve((size_t)N * C_DIM * sizeof(float));     // 51.2 MB
    float*    s1      = (float*)   carve((size_t)N_HEADS * N * sizeof(float));
    float*    s2      = (float*)   carve((size_t)N_HEADS * N * sizeof(float));
    int*      row_ptr = (int*)     carve((size_t)(N + 1) * sizeof(int));
    (void)ws_size;

    // 1) h -> f16
    {
        int total = N * IN_DIM;
        cvt_h_kernel<<<(total + 255) / 256, 256, 0, stream>>>(h, hh, total);
    }
    // 2) Wt (f16, col-major over concat heads)
    build_wt_kernel<<<C_DIM, IN_DIM, 0, stream>>>(W, Wt);
    // 3) z = hh @ Wcat via WMMA
    {
        dim3 grid((N + 31) / 32, C_DIM / 64);
        gemm_wmma_kernel<<<grid, 256, 0, stream>>>(hh, Wt, z, N);
    }
    // 4) per-node attention scalars
    {
        int total = N * N_HEADS;
        attn_scal_kernel<<<(total + 255) / 256, 256, 0, stream>>>(z, a, s1, s2, N);
    }
    // 5) CSR row pointers
    rowptr_kernel<<<(N + 1 + 255) / 256, 256, 0, stream>>>(dst, E, N, row_ptr);
    // 6) softmax-weighted aggregation, one wave per (node, head)
    gat_aggregate_kernel<<<N, 128, 0, stream>>>(z, s1, s2, src, row_ptr, out, N);
}
